// MCoreAttention_29686813950454
// MI455X (gfx1250) — matrix operand order repaired and checked by reading, not compiled, not run on hardware
//
#include <hip/hip_runtime.h>

// ---------------------------------------------------------------------------
// Types for CDNA5 WMMA (wave32, 16x16x32 bf16 -> f32 accumulate)
// ---------------------------------------------------------------------------
typedef __bf16 v16bf __attribute__((ext_vector_type(16)));
typedef __bf16 v8bf  __attribute__((ext_vector_type(8)));
typedef float  v8f   __attribute__((ext_vector_type(8)));

union V16U { v16bf v; v8bf h[2]; };

#define NUM_HEADS   32
#define NUM_KV      8
#define HEAD_DIM    64
#define SEQ         2048
#define HID         2048
#define QKV_N       3072   // (32 + 2*8) * 64
#define BATCH       2

// float -> bf16 round-to-nearest-even, stored as ushort
__device__ __forceinline__ unsigned short f2bf(float f) {
    unsigned int u = __float_as_uint(f);
    u += 0x7FFFu + ((u >> 16) & 1u);
    return (unsigned short)(u >> 16);
}

// CDNA5 async global->LDS copy (ASYNCcnt-tracked, no VGPR round trip).
// GVS addressing: 64-bit SGPR base + 32-bit per-lane VGPR byte offset.
__device__ __forceinline__ void async_copy_b128(unsigned lds_addr,
                                                const void* sbase,
                                                unsigned voff) {
    asm volatile("global_load_async_to_lds_b128 %0, %1, %2"
                 :
                 : "v"(lds_addr), "v"(voff),
                   "s"((unsigned long long)(__SIZE_TYPE__)sbase)
                 : "memory");
}

// ASYNCcnt completes in order: waiting for <= n leaves the newest n in flight.
#define WAIT_ASYNC(n) asm volatile("s_wait_asynccnt " #n ::: "memory")

__device__ __forceinline__ unsigned lds_addr32(const void* p) {
    return (unsigned)(__SIZE_TYPE__)p;
}

// ---------------------------------------------------------------------------
// Kernel 1: fp32 -> bf16 (storage ushort)
// ---------------------------------------------------------------------------
__global__ __launch_bounds__(256)
void cvt_f32_bf16_kernel(const float* __restrict__ in, unsigned short* __restrict__ out, int n) {
    int i = blockIdx.x * blockDim.x + threadIdx.x;
    int stride = gridDim.x * blockDim.x;
    for (; i < n; i += stride) out[i] = f2bf(in[i]);
}

// ---------------------------------------------------------------------------
// Kernel 2: GEMM  C[M,N] = A[M,K] * W[N,K]^T + bias[N]   (A,W bf16; C f32)
// 256 threads = 8 waves (2 M x 4 N); block tile 128(M) x 256(N).
// K staged in LDS in 64-deep double-buffered tiles via async copies, with the
// next tile's asyncs issued before waiting on the current (in-order ASYNCcnt).
// Wave computes 64x64 = 4x4 accumulators of 16x16 -> 512 B LDS per WMMA.
// Fragment layouts per CDNA5 ISA 7.12.2 (16-bit A 16x32, B 32x16).
// ---------------------------------------------------------------------------
#define GSTR 72   // padded LDS row stride (ushorts): 144 B, 16B-aligned chunks

__global__ __launch_bounds__(256)
void gemm_bf16_wmma_kernel(const unsigned short* __restrict__ A,
                           const unsigned short* __restrict__ W,
                           const float* __restrict__ bias,
                           float* __restrict__ C,
                           int M, int N, int K) {
    __shared__ unsigned short As[2][128 * GSTR];   // 36 KB
    __shared__ unsigned short Bs[2][256 * GSTR];   // 72 KB

    const int lane  = threadIdx.x & 31;
    const int wave  = threadIdx.x >> 5;
    const int lr    = lane & 15;       // row/col within 16
    const int hs    = lane >> 4;       // k-half select
    const int waveM = wave >> 2;       // 0..1 -> 64 rows each
    const int waveN = wave & 3;        // 0..3 -> 64 cols each
    const int mblk  = blockIdx.y * 128;
    const int nblk  = blockIdx.x * 256;

    v8f acc[4][4];
#pragma unroll
    for (int mi = 0; mi < 4; ++mi)
#pragma unroll
        for (int ni = 0; ni < 4; ++ni) acc[mi][ni] = (v8f)(0.0f);

    const int NKB = K / 64;

    // Cooperative async prefetch: A tile 128x64 (1024 chunks) + B tile 256x64
    // (2048 chunks); 256 threads -> 12 async instructions per wave.
    auto prefetch = [&](int buf, int k0) {
#pragma unroll
        for (int i = 0; i < 4; ++i) {
            int seg   = i * 256 + threadIdx.x;   // 0..1023
            int row   = seg >> 3;                // 0..127
            int chunk = seg & 7;                 // 16B chunk within 64 k
            unsigned avoff = (unsigned)(((mblk + row) * (size_t)K + k0) * 2) + chunk * 16;
            async_copy_b128(lds_addr32(&As[buf][row * GSTR + chunk * 8]), A, avoff);
        }
#pragma unroll
        for (int i = 0; i < 8; ++i) {
            int seg   = i * 256 + threadIdx.x;   // 0..2047
            int row   = seg >> 3;                // 0..255
            int chunk = seg & 7;
            unsigned bvoff = (unsigned)(((nblk + row) * (size_t)K + k0) * 2) + chunk * 16;
            async_copy_b128(lds_addr32(&Bs[buf][row * GSTR + chunk * 8]), W, bvoff);
        }
    };

    prefetch(0, 0);

    for (int kb = 0; kb < NKB; ++kb) {
        const int buf = kb & 1;
        if (kb + 1 < NKB) {
            prefetch(buf ^ 1, (kb + 1) * 64);  // issue next tile first
            WAIT_ASYNC(12);                    // ...then wait only for current
        } else {
            WAIT_ASYNC(0);
        }
        __syncthreads();        // everyone's share of tile `buf` visible

        const unsigned short* as = &As[buf][0];
        const unsigned short* bs = &Bs[buf][0];
#pragma unroll
        for (int ks = 0; ks < 2; ++ks) {      // two 32-deep WMMA k-steps
            v16bf a[4];
#pragma unroll
            for (int mi = 0; mi < 4; ++mi) {
                const unsigned short* ap =
                    as + (waveM * 64 + mi * 16 + lr) * GSTR + ks * 32 + hs * 8;
                V16U u;
                u.h[0] = *(const v8bf*)(const void*)ap;
                u.h[1] = *(const v8bf*)(const void*)(ap + 16);
                a[mi] = u.v;
            }
            v16bf b[4];
#pragma unroll
            for (int ni = 0; ni < 4; ++ni) {
                const unsigned short* bp =
                    bs + (waveN * 64 + ni * 16 + lr) * GSTR + ks * 32 + hs * 8;
                V16U u;
                u.h[0] = *(const v8bf*)(const void*)bp;
                u.h[1] = *(const v8bf*)(const void*)(bp + 16);
                b[ni] = u.v;
            }
#pragma unroll
            for (int mi = 0; mi < 4; ++mi)
#pragma unroll
                for (int ni = 0; ni < 4; ++ni)
                    acc[mi][ni] = __builtin_amdgcn_wmma_f32_16x16x32_bf16(
                        false, a[mi], false, b[ni], (short)0, acc[mi][ni], false, false);
        }
        __syncthreads();        // all reads of `buf` done before it is refilled
    }

    // Epilogue: C layout — vgpr v: row v (lanes 0-15) / v+8 (lanes 16-31), col = lane%16
#pragma unroll
    for (int mi = 0; mi < 4; ++mi)
#pragma unroll
        for (int ni = 0; ni < 4; ++ni) {
            int c = nblk + waveN * 64 + ni * 16 + lr;
            float bb = bias[c];
#pragma unroll
            for (int v = 0; v < 8; ++v) {
                int r = mblk + waveM * 64 + mi * 16 + v + 8 * hs;
                C[(size_t)r * N + c] = acc[mi][ni][v] + bb;
            }
        }
}

// ---------------------------------------------------------------------------
// Kernel 3: RoPE + repack  qkv[B,S,3072] f32 -> Q[B,32,S,64], K[B,8,S,64],
// V[B,8,S,64] (bf16). qkv channel layout: kv-group kh: q slots 0..3, k=4, v=5.
// ---------------------------------------------------------------------------
__global__ __launch_bounds__(256)
void rope_pack_kernel(const float* __restrict__ qkv, const int* __restrict__ pos_ids,
                      unsigned short* __restrict__ Qo, unsigned short* __restrict__ Ko,
                      unsigned short* __restrict__ Vo) {
    const int s = blockIdx.x;
    const int b = blockIdx.y;
    const float pos = (float)pos_ids[s];
    const float* row = qkv + ((size_t)b * SEQ + s) * QKV_N;
    const float nlog_theta = -11.918390573078392f; // -ln(150000)

    for (int t = threadIdx.x; t < 40 * 32; t += 256) {
        int hh = t >> 5;          // 0..39 : 0..31 q heads, 32..39 k heads
        int i  = t & 31;
        float inv = __expf(((float)i * (1.0f / 32.0f)) * nlog_theta);
        float ang = pos * inv;
        float c = __cosf(ang), sn = __sinf(ang);
        if (hh < NUM_HEADS) {
            int off = (hh >> 2) * 384 + (hh & 3) * 64;
            float x0 = row[off + i], x1 = row[off + i + 32];
            size_t base = (((size_t)b * NUM_HEADS + hh) * SEQ + s) * HEAD_DIM;
            Qo[base + i]      = f2bf(x0 * c - x1 * sn);
            Qo[base + i + 32] = f2bf(x1 * c + x0 * sn);
        } else {
            int kh = hh - NUM_HEADS;
            int off = kh * 384 + 256;
            float x0 = row[off + i], x1 = row[off + i + 32];
            size_t base = (((size_t)b * NUM_KV + kh) * SEQ + s) * HEAD_DIM;
            Ko[base + i]      = f2bf(x0 * c - x1 * sn);
            Ko[base + i + 32] = f2bf(x1 * c + x0 * sn);
        }
    }
    for (int t = threadIdx.x; t < NUM_KV * HEAD_DIM; t += 256) {
        int kh = t >> 6, d = t & 63;
        size_t base = (((size_t)b * NUM_KV + kh) * SEQ + s) * HEAD_DIM;
        Vo[base + d] = f2bf(row[kh * 384 + 320 + d]);
    }
}

// ---------------------------------------------------------------------------
// Kernel 4: GQA flash attention (non-causal, full softmax).
// Grid: (S/64, 32 heads, B). Block: 128 thr = 4 waves; wave owns 16 q rows.
// K tile [64 keys][64 d] async-copied to LDS; V^T tile built by scatter.
// ---------------------------------------------------------------------------
#define LDS_STRIDE 72   // 64 + 8 pad (ushort units)

__global__ __launch_bounds__(128)
void flash_attn_kernel(const unsigned short* __restrict__ Q,
                       const unsigned short* __restrict__ Kg,
                       const unsigned short* __restrict__ Vg,
                       unsigned short* __restrict__ Out) {
    __shared__ unsigned short Kt[64 * LDS_STRIDE];
    __shared__ unsigned short Vt[64 * LDS_STRIDE];          // transposed [d][key]
    __shared__ unsigned short Pl[4 * 16 * LDS_STRIDE];      // per-wave P tiles

    const int lane = threadIdx.x & 31;
    const int wave = threadIdx.x >> 5;
    const int lr = lane & 15;
    const int hs = lane >> 4;
    const int qt = blockIdx.x, h = blockIdx.y, b = blockIdx.z;
    const int kvh = h >> 2;   // 4 q heads per kv head

    const unsigned short* qbase = Q + (((size_t)b * NUM_HEADS + h) * SEQ + qt * 64 + wave * 16) * HEAD_DIM;
    const unsigned short* kbase = Kg + ((size_t)b * NUM_KV + kvh) * SEQ * HEAD_DIM;
    const unsigned short* vbase = Vg + ((size_t)b * NUM_KV + kvh) * SEQ * HEAD_DIM;

    // Q fragments (16 rows x 64 dim -> two 16x32 A fragments), resident in VGPRs
    v16bf aq[2];
#pragma unroll
    for (int kb = 0; kb < 2; ++kb) {
        const unsigned short* p = qbase + (size_t)lr * HEAD_DIM + kb * 32 + hs * 8;
        V16U u;
        u.h[0] = *(const v8bf*)(const void*)p;
        u.h[1] = *(const v8bf*)(const void*)(p + 16);
        aq[kb] = u.v;
    }

    v8f o[4];
    float mrun[8], lrun[8];
#pragma unroll
    for (int dt = 0; dt < 4; ++dt) o[dt] = (v8f)(0.0f);
#pragma unroll
    for (int v = 0; v < 8; ++v) { mrun[v] = -1e30f; lrun[v] = 0.0f; }

    const float sc = 0.125f; // 1/sqrt(64)

    for (int j = 0; j < SEQ / 64; ++j) {
        __syncthreads();   // previous iteration's LDS reads done before overwrite
        {
            // K tile: 64 rows x 128 B = 512 16B chunks, async to LDS
#pragma unroll
            for (int i = 0; i < 4; ++i) {
                int seg = i * 128 + threadIdx.x;
                int row = seg >> 3, chunk = seg & 7;
                unsigned voff = (unsigned)((j * 64 + row) * (HEAD_DIM * 2)) + chunk * 16;
                async_copy_b128(lds_addr32(&Kt[row * LDS_STRIDE + chunk * 8]), kbase, voff);
            }
            // V^T tile: scalar transpose scatter (bank-strided b16 stores)
            int r = threadIdx.x >> 1, half = threadIdx.x & 1;
            const unsigned short* vsrc = vbase + (size_t)(j * 64 + r) * HEAD_DIM + half * 32;
#pragma unroll
            for (int d = 0; d < 32; ++d)
                Vt[(half * 32 + d) * LDS_STRIDE + r] = vsrc[d];
        }
        WAIT_ASYNC(0);
        __syncthreads();

        // ---- scores: S(16x64) = Q(16x64) * K^T ----
        v8f s[4];
#pragma unroll
        for (int ni = 0; ni < 4; ++ni) {
            s[ni] = (v8f)(0.0f);
#pragma unroll
            for (int kb = 0; kb < 2; ++kb) {
                const unsigned short* kp =
                    &Kt[(ni * 16 + lr) * LDS_STRIDE + kb * 32 + hs * 8];
                V16U u;
                u.h[0] = *(const v8bf*)(const void*)kp;
                u.h[1] = *(const v8bf*)(const void*)(kp + 16);
                s[ni] = __builtin_amdgcn_wmma_f32_16x16x32_bf16(
                    false, aq[kb], false, u.v, (short)0, s[ni], false, false);
            }
        }

        // ---- online softmax (row = vgpr v + 8*hs, cols across lanes 0..15) ----
#pragma unroll
        for (int v = 0; v < 8; ++v) {
            float mx = fmaxf(fmaxf(s[0][v], s[1][v]), fmaxf(s[2][v], s[3][v])) * sc;
#pragma unroll
            for (int msk = 1; msk < 16; msk <<= 1)
                mx = fmaxf(mx, __shfl_xor(mx, msk, 32));
            float mnew = fmaxf(mrun[v], mx);
            float corr = __expf(mrun[v] - mnew);
            mrun[v] = mnew;
            float rsum = 0.0f;
#pragma unroll
            for (int ni = 0; ni < 4; ++ni) {
                float p = __expf(s[ni][v] * sc - mnew);
                s[ni][v] = p;
                rsum += p;
            }
#pragma unroll
            for (int msk = 1; msk < 16; msk <<= 1)
                rsum += __shfl_xor(rsum, msk, 32);
            lrun[v] = lrun[v] * corr + rsum;
#pragma unroll
            for (int dt = 0; dt < 4; ++dt) o[dt][v] *= corr;
        }

        // ---- P -> LDS (row major) then reload as A fragments ----
        unsigned short* pw = &Pl[wave * 16 * LDS_STRIDE];
#pragma unroll
        for (int ni = 0; ni < 4; ++ni)
#pragma unroll
            for (int v = 0; v < 8; ++v)
                pw[(v + 8 * hs) * LDS_STRIDE + ni * 16 + lr] = f2bf(s[ni][v]);

        v16bf ap[2];
#pragma unroll
        for (int kb = 0; kb < 2; ++kb) {
            const unsigned short* pp = pw + (size_t)lr * LDS_STRIDE + kb * 32 + hs * 8;
            V16U u;
            u.h[0] = *(const v8bf*)(const void*)pp;
            u.h[1] = *(const v8bf*)(const void*)(pp + 16);
            ap[kb] = u.v;
        }

        // ---- O += P(16x64) * V(64x64), V^T in LDS gives contiguous B frags ----
#pragma unroll
        for (int dt = 0; dt < 4; ++dt)
#pragma unroll
            for (int kb = 0; kb < 2; ++kb) {
                const unsigned short* vp =
                    &Vt[(dt * 16 + lr) * LDS_STRIDE + kb * 32 + hs * 8];
                V16U u;
                u.h[0] = *(const v8bf*)(const void*)vp;
                u.h[1] = *(const v8bf*)(const void*)(vp + 16);
                o[dt] = __builtin_amdgcn_wmma_f32_16x16x32_bf16(
                    false, ap[kb], false, u.v, (short)0, o[dt], false, false);
            }
    }

    // ---- normalize and store: Out[B,S, h*64 + d] bf16 ----
#pragma unroll
    for (int dt = 0; dt < 4; ++dt)
#pragma unroll
        for (int v = 0; v < 8; ++v) {
            int tok = qt * 64 + wave * 16 + v + 8 * hs;
            float val = o[dt][v] / lrun[v];
            Out[((size_t)b * SEQ + tok) * (NUM_HEADS * HEAD_DIM) + h * 64 + dt * 16 + lr] = f2bf(val);
        }
}

// ---------------------------------------------------------------------------
// Host launch
// ---------------------------------------------------------------------------
extern "C" void kernel_launch(void* const* d_in, const int* in_sizes, int n_in,
                              void* d_out, int out_size, void* d_ws, size_t ws_size,
                              hipStream_t stream) {
    const float* hidden = (const float*)d_in[0];   // [2,2048,2048]
    const int*   pos    = (const int*)d_in[1];     // [1,2048]
    const float* qkv_w  = (const float*)d_in[2];   // [3072,2048]
    const float* qkv_b  = (const float*)d_in[3];   // [3072]
    const float* o_w    = (const float*)d_in[4];   // [2048,2048]
    const float* o_b    = (const float*)d_in[5];   // [2048]
    float* out = (float*)d_out;                    // [2,2048,2048]

    const size_t M = (size_t)BATCH * SEQ;          // 4096

    char* p = (char*)d_ws;
    unsigned short* hid_bf  = (unsigned short*)p;  p += M * HID * 2;
    unsigned short* qkvw_bf = (unsigned short*)p;  p += (size_t)QKV_N * HID * 2;
    unsigned short* ow_bf   = (unsigned short*)p;  p += (size_t)HID * HID * 2;
    float*          qkv_f   = (float*)p;           p += M * QKV_N * 4;
    unsigned short* q_bf    = (unsigned short*)p;  p += (size_t)BATCH * NUM_HEADS * SEQ * HEAD_DIM * 2;
    unsigned short* k_bf    = (unsigned short*)p;  p += (size_t)BATCH * NUM_KV * SEQ * HEAD_DIM * 2;
    unsigned short* v_bf    = (unsigned short*)p;  p += (size_t)BATCH * NUM_KV * SEQ * HEAD_DIM * 2;
    unsigned short* ao_bf   = (unsigned short*)p;  p += M * HID * 2;

    // 1) downconvert operands to bf16
    cvt_f32_bf16_kernel<<<4096, 256, 0, stream>>>(hidden, hid_bf, (int)(M * HID));
    cvt_f32_bf16_kernel<<<4096, 256, 0, stream>>>(qkv_w, qkvw_bf, QKV_N * HID);
    cvt_f32_bf16_kernel<<<4096, 256, 0, stream>>>(o_w, ow_bf, HID * HID);

    // 2) QKV projection: [4096,2048] x [3072,2048]^T + qkv_b -> f32
    gemm_bf16_wmma_kernel<<<dim3(QKV_N / 256, M / 128), 256, 0, stream>>>(
        hid_bf, qkvw_bf, qkv_b, qkv_f, (int)M, QKV_N, HID);

    // 3) RoPE + repack to [B,H,S,D] bf16
    rope_pack_kernel<<<dim3(SEQ, BATCH), 256, 0, stream>>>(qkv_f, pos, q_bf, k_bf, v_bf);

    // 4) GQA flash attention
    flash_attn_kernel<<<dim3(SEQ / 64, NUM_HEADS, BATCH), 128, 0, stream>>>(
        q_bf, k_bf, v_bf, ao_bf);

    // 5) output projection: [4096,2048] x [2048,2048]^T + o_b -> d_out f32
    gemm_bf16_wmma_kernel<<<dim3(HID / 256, M / 128), 256, 0, stream>>>(
        ao_bf, ow_bf, o_b, out, (int)M, HID, HID);
}
